// LSTMAttnSeq2Seq_87454124081538
// MI455X (gfx1250) — matrix operand order repaired and checked by reading, not compiled, not run on hardware
//
#include <hip/hip_runtime.h>
#include <hip/hip_bf16.h>
#include <stdint.h>

// Problem constants (from reference)
#define BB   32      // batch
#define HH   512     // hidden / ninp / nattn
#define LL   64      // src len
#define TLm1 63      // trg len - 1
#define VV   32000   // vocab
#define G4   2048    // 4*H (gates)
#define KC   512     // GEMM K-chunk staged in LDS (all K are multiples of 512)

typedef __attribute__((ext_vector_type(16))) __bf16 v16bf;
typedef __attribute__((ext_vector_type(8)))  __bf16 v8bf;
typedef __attribute__((ext_vector_type(8)))  float  v8f;

// ---------------- elementwise / setup kernels ----------------

__global__ void k_f32_to_bf16(const float* __restrict__ s, __hip_bfloat16* __restrict__ d, int n) {
  int i = blockIdx.x * blockDim.x + threadIdx.x;
  if (i < n) d[i] = __float2bfloat16(s[i]);
}

// Wcat[n, 0:Kin] = W_ih[n,:]; Wcat[n, Kin:Kin+H] = W_hh[n,:]  (bf16)
__global__ void k_pack_lstm(const float* __restrict__ Wih, const float* __restrict__ Whh,
                            __hip_bfloat16* __restrict__ Wcat, int Kin, int n) {
  int i = blockIdx.x * blockDim.x + threadIdx.x;
  if (i >= n) return;
  int ktot = Kin + HH;
  int r = i / ktot, k = i - r * ktot;
  float v = (k < Kin) ? Wih[(size_t)r * Kin + k] : Whh[(size_t)r * HH + (k - Kin)];
  Wcat[i] = __float2bfloat16(v);
}

__global__ void k_bsum(const float* __restrict__ a, const float* __restrict__ b,
                       float* __restrict__ o, int n) {
  int i = blockIdx.x * blockDim.x + threadIdx.x;
  if (i < n) o[i] = a[i] + b[i];
}

// dst[r, 0:width] (row stride dstLd, bf16) = table[tok[r], :] (f32)
__global__ void k_gather_embed(const int* __restrict__ tok, const float* __restrict__ tab,
                               __hip_bfloat16* __restrict__ dst, int rows, int width, int dstLd) {
  int i = blockIdx.x * blockDim.x + threadIdx.x;
  if (i >= rows * width) return;
  int r = i / width, j = i - r * width;
  dst[(size_t)r * dstLd + j] = __float2bfloat16(tab[(size_t)tok[r] * width + j]);
}

__global__ void k_copy_rows_bf16(const __hip_bfloat16* __restrict__ src, int sLd,
                                 __hip_bfloat16* __restrict__ dst, int dLd, int rows, int width) {
  int i = blockIdx.x * blockDim.x + threadIdx.x;
  if (i >= rows * width) return;
  int r = i / width, j = i - r * width;
  dst[(size_t)r * dLd + j] = src[(size_t)r * sLd + j];
}

__global__ void k_zero_f32(float* __restrict__ p, size_t n) {
  size_t i = (size_t)blockIdx.x * blockDim.x + threadIdx.x;
  if (i < n) p[i] = 0.0f;
}

// ---------------- WMMA GEMM: C[M,N] = A[M,K] * B[N,K]^T (+bias[N]) ----------------
// A,B bf16 row-major (leading dims over K), C f32 (ldc over N).
// Block = 256 threads (8 waves) owns a 32(M) x 256(N) tile of C.
// A strip (32 x KC) is staged into LDS once per K-chunk with ASYNC global->LDS
// copies (ASYNCcnt path); waves read A fragments from LDS (ds_load_b128) and
// stream B from global/L2. Each wave: 1 m-tile x 4 n-tiles, A fragment reused 4x.
// Requires: M%32==0, N%256==0, K%KC==0, lda/ldb multiples of 16 elems.
// Grid is exact -> every thread participates in barriers / async copies (EXEC all 1s).
__global__ void gemm_bf16_nt(const __hip_bfloat16* __restrict__ A, int lda,
                             const __hip_bfloat16* __restrict__ Bm, int ldb,
                             float* __restrict__ C, int ldc,
                             const float* __restrict__ bias,
                             int K) {
  __shared__ __align__(128) __hip_bfloat16 As[32 * KC];
  const int tid  = threadIdx.x;
  const int lane = tid & 31;
  const int wave = tid >> 5;            // 0..7
  const int mtl  = wave & 1;            // local m-tile (0..1)
  const int ngl  = wave >> 1;           // local n-group (0..3), 64 cols each
  const int half = lane >> 4;           // K-half selector (contiguous k order: k = 16*half + s)
  const int l16  = lane & 15;
  const int mrow0 = blockIdx.y * 32;
  const int gng   = blockIdx.x * 4 + ngl;

  const __bf16* Ag = (const __bf16*)(const void*)A + (size_t)mrow0 * lda;
  const __bf16* Br = (const __bf16*)(const void*)Bm + (size_t)(gng * 64 + l16) * ldb + half * 16;
  const __bf16* Al = (const __bf16*)(const void*)&As[(mtl * 16 + l16) * KC];

  v8f acc0 = {}, acc1 = {}, acc2 = {}, acc3 = {};
  for (int kc = 0; kc < K; kc += KC) {
    __syncthreads();   // previous chunk fully consumed
    // --- async stage A[mrow0:+32, kc:+KC) -> LDS (row-major, stride KC) ---
    // 32*KC bf16 = 32KB = 2048 x 16B units; 8 units per thread.
    for (int u = tid * 8; u < 32 * KC; u += 256 * 8) {
      const int row = u / KC, col = u - (u / KC) * KC;
      const __bf16* g = Ag + (size_t)row * lda + kc + col;
      unsigned lds = (unsigned)(uintptr_t)&As[u];            // LDS offset (low 32 bits)
      unsigned long long ga = (unsigned long long)(uintptr_t)g;
      asm volatile("global_load_async_to_lds_b128 %0, %1, off"
                   :: "v"(lds), "v"(ga) : "memory");
    }
    asm volatile("s_wait_asynccnt 0x0" ::: "memory");
    __syncthreads();
    // --- compute over the chunk ---
#pragma unroll 4
    for (int k0 = 0; k0 < KC; k0 += 32) {
      // A 16x32 fragment from LDS: lane<16 holds K {0..15}, lane>=16 {16..31}
      v8bf alo = *(const v8bf*)(Al + k0 + half * 16);
      v8bf ahi = *(const v8bf*)(Al + k0 + half * 16 + 8);
      v16bf a;
#pragma unroll
      for (int i = 0; i < 8; ++i) { a[i] = alo[i]; a[i + 8] = ahi[i]; }
      // B 32x16 fragments from global: lane%16 = column (row of Bm), half selects K-half
      const __bf16* Bk = Br + kc + k0;
      v16bf b0 = *(const v16bf*)(Bk);
      v16bf b1 = *(const v16bf*)(Bk + (size_t)16 * ldb);
      v16bf b2 = *(const v16bf*)(Bk + (size_t)32 * ldb);
      v16bf b3 = *(const v16bf*)(Bk + (size_t)48 * ldb);
      __builtin_prefetch((const void*)(Bk + 128), 0, 1);
      acc0 = __builtin_amdgcn_wmma_f32_16x16x32_bf16(false, a, false, b0, (short)0, acc0, false, false);
      acc1 = __builtin_amdgcn_wmma_f32_16x16x32_bf16(false, a, false, b1, (short)0, acc1, false, false);
      acc2 = __builtin_amdgcn_wmma_f32_16x16x32_bf16(false, a, false, b2, (short)0, acc2, false, false);
      acc3 = __builtin_amdgcn_wmma_f32_16x16x32_bf16(false, a, false, b3, (short)0, acc3, false, false);
    }
  }
  // C/D layout: VGPR v, lanes0-15 -> M=v, lanes16-31 -> M=v+8; N = lane%16
  const int row0 = mrow0 + mtl * 16 + half * 8;
  v8f accs[4] = {acc0, acc1, acc2, acc3};
#pragma unroll
  for (int nn = 0; nn < 4; ++nn) {
    const int col = gng * 64 + nn * 16 + l16;
    const float bv = bias ? bias[col] : 0.0f;
#pragma unroll
    for (int v = 0; v < 8; ++v)
      C[(size_t)(row0 + v) * ldc + col] = accs[nn][v] + bv;
  }
}

// ---------------- LSTM gate nonlinearity ----------------
__global__ void k_lstm_cell(const float* __restrict__ z, const float* __restrict__ bsum,
                            float* __restrict__ c,
                            __hip_bfloat16* __restrict__ hb0, int ld0,
                            __hip_bfloat16* __restrict__ hb1, int ld1,
                            float* __restrict__ hf, int ldf) {
  int i = blockIdx.x * blockDim.x + threadIdx.x;
  if (i >= BB * HH) return;
  int b = i >> 9, j = i & (HH - 1);
  const float* zr = z + (size_t)b * G4;
  float zi = zr[j]          + bsum[j];
  float zf = zr[HH + j]     + bsum[HH + j];
  float zg = zr[2 * HH + j] + bsum[2 * HH + j];
  float zo = zr[3 * HH + j] + bsum[3 * HH + j];
  float si = 1.0f / (1.0f + __expf(-zi));
  float sf = 1.0f / (1.0f + __expf(-zf));
  float so = 1.0f / (1.0f + __expf(-zo));
  float cn = sf * c[i] + si * tanhf(zg);
  float hn = so * tanhf(cn);
  c[i] = cn;
  __hip_bfloat16 hb = __float2bfloat16(hn);
  hb0[(size_t)b * ld0 + j] = hb;
  if (hb1) hb1[(size_t)b * ld1 + j] = hb;
  if (hf)  hf[(size_t)b * ldf + j]  = hn;
}

// ---------------- fused attention: scores + mask + softmax + context ----------------
__global__ void k_attn(const float* __restrict__ Wenc, const float* __restrict__ q,
                       const float* __restrict__ enc_o, const float* __restrict__ W3,
                       const float* __restrict__ b3, const int* __restrict__ length,
                       __hip_bfloat16* __restrict__ ct, int ldct) {
  __shared__ float qs[HH];
  __shared__ float red[256];
  __shared__ float tl[LL];
  __shared__ float al[LL];
  const int tid = threadIdx.x;
  const int b   = blockIdx.x;
  for (int i = tid; i < HH; i += 256) qs[i] = q[(size_t)b * HH + i];
  __syncthreads();
  const float b3v = b3[0];
  const int myLen = length[b];
  for (int l = 0; l < LL; ++l) {
    const float* we = Wenc + ((size_t)l * BB + b) * HH;
    float local = 0.0f;
    for (int i = tid; i < HH; i += 256) local += W3[i] * tanhf(we[i] + qs[i]);
    red[tid] = local; __syncthreads();
    for (int off = 128; off > 0; off >>= 1) {
      if (tid < off) red[tid] += red[tid + off];
      __syncthreads();
    }
    if (tid == 0) tl[l] = (myLen < l) ? -INFINITY : (red[0] + b3v);
    __syncthreads();
  }
  float m = -INFINITY;
  for (int l = tid; l < LL; l += 256) m = fmaxf(m, tl[l]);
  red[tid] = m; __syncthreads();
  for (int off = 128; off > 0; off >>= 1) {
    if (tid < off) red[tid] = fmaxf(red[tid], red[tid + off]);
    __syncthreads();
  }
  const float mx = red[0]; __syncthreads();
  float s = 0.0f;
  for (int l = tid; l < LL; l += 256) { float e = __expf(tl[l] - mx); al[l] = e; s += e; }
  red[tid] = s; __syncthreads();
  for (int off = 128; off > 0; off >>= 1) {
    if (tid < off) red[tid] += red[tid + off];
    __syncthreads();
  }
  const float inv = 1.0f / red[0];
  __syncthreads();
  for (int h = tid; h < HH; h += 256) {
    float acc = 0.0f;
    for (int l = 0; l < LL; ++l) acc += al[l] * enc_o[((size_t)l * BB + b) * HH + h];
    ct[(size_t)b * ldct + h] = __float2bfloat16(acc * inv);
  }
}

// ---------------- host driver ----------------

static inline void launch_gemm(const void* A, int lda, const void* Bm, int ldb,
                               float* C, int ldc, const float* bias,
                               int M, int N, int K, hipStream_t stream) {
  dim3 grid(N / 256, M / 32);   // exact: M%32==0, N%256==0 for all call sites
  gemm_bf16_nt<<<grid, 256, 0, stream>>>((const __hip_bfloat16*)A, lda,
                                         (const __hip_bfloat16*)Bm, ldb,
                                         C, ldc, bias, K);
}

extern "C" void kernel_launch(void* const* d_in, const int* in_sizes, int n_in,
                              void* d_out, int out_size, void* d_ws, size_t ws_size,
                              hipStream_t stream) {
  (void)in_sizes; (void)n_in; (void)out_size; (void)ws_size;
  const int*   x   = (const int*)d_in[0];
  const int*   y   = (const int*)d_in[1];
  const int*   len = (const int*)d_in[2];
  const float* src_embed = (const float*)d_in[3];
  const float* trg_embed = (const float*)d_in[4];
  const float* b_out     = (const float*)d_in[5];
  const float* eWih[2] = {(const float*)d_in[6],  (const float*)d_in[10]};
  const float* eWhh[2] = {(const float*)d_in[7],  (const float*)d_in[11]};
  const float* ebih[2] = {(const float*)d_in[8],  (const float*)d_in[12]};
  const float* ebhh[2] = {(const float*)d_in[9],  (const float*)d_in[13]};
  const float* dWih[2] = {(const float*)d_in[14], (const float*)d_in[18]};
  const float* dWhh[2] = {(const float*)d_in[15], (const float*)d_in[19]};
  const float* dbih[2] = {(const float*)d_in[16], (const float*)d_in[20]};
  const float* dbhh[2] = {(const float*)d_in[17], (const float*)d_in[21]};
  const float* W1 = (const float*)d_in[22];
  const float* b1 = (const float*)d_in[23];
  const float* W2 = (const float*)d_in[24];
  const float* b2 = (const float*)d_in[25];
  const float* W3 = (const float*)d_in[26];
  const float* b3 = (const float*)d_in[27];
  float* out = (float*)d_out;

  // ---- workspace carving (256B aligned slabs) ----
  char* p = (char*)d_ws;
  auto carve = [&](size_t bytes) -> void* {
    void* r = (void*)p; p += (bytes + 255) & ~(size_t)255; return r;
  };
  __hip_bfloat16* trg_bf  = (__hip_bfloat16*)carve((size_t)VV * HH * 2);   // L2-resident logits weight
  __hip_bfloat16* Wc_e0   = (__hip_bfloat16*)carve((size_t)G4 * 1024 * 2); // [Wih|Whh] enc l0, K=1024
  __hip_bfloat16* Wc_e1   = (__hip_bfloat16*)carve((size_t)G4 * 1024 * 2);
  __hip_bfloat16* Wc_d0   = (__hip_bfloat16*)carve((size_t)G4 * 1536 * 2); // dec l0, K=1536
  __hip_bfloat16* Wc_d1   = (__hip_bfloat16*)carve((size_t)G4 * 1024 * 2);
  float* bs_e0 = (float*)carve(G4 * 4);
  float* bs_e1 = (float*)carve(G4 * 4);
  float* bs_d0 = (float*)carve(G4 * 4);
  float* bs_d1 = (float*)carve(G4 * 4);
  __hip_bfloat16* W1bf = (__hip_bfloat16*)carve((size_t)HH * HH * 2);
  __hip_bfloat16* W2bf = (__hip_bfloat16*)carve((size_t)HH * HH * 2);
  __hip_bfloat16* A0   = (__hip_bfloat16*)carve((size_t)LL * BB * HH * 2); // gathered src emb
  float* enc_o            = (float*)carve((size_t)LL * BB * HH * 4);
  __hip_bfloat16* enc_obf = (__hip_bfloat16*)carve((size_t)LL * BB * HH * 2);
  float* Wenc             = (float*)carve((size_t)LL * BB * HH * 4);
  char* arena0 = p;  // zeroed state arena
  float* c_e0 = (float*)carve((size_t)BB * HH * 4);
  float* c_e1 = (float*)carve((size_t)BB * HH * 4);
  float* c_d0 = (float*)carve((size_t)BB * HH * 4);
  float* c_d1 = (float*)carve((size_t)BB * HH * 4);
  __hip_bfloat16* xh0  = (__hip_bfloat16*)carve((size_t)BB * 1024 * 2); // [x|h] enc0
  __hip_bfloat16* xh1  = (__hip_bfloat16*)carve((size_t)BB * 1024 * 2); // [h0|h1] enc1
  __hip_bfloat16* xhd0 = (__hip_bfloat16*)carve((size_t)BB * 1536 * 2); // [emb|ct|h] dec0
  __hip_bfloat16* xhd1 = (__hip_bfloat16*)carve((size_t)BB * 1024 * 2); // [h0|h1] dec1
  float* zbuf = (float*)carve((size_t)BB * G4 * 4);
  float* qbuf = (float*)carve((size_t)BB * HH * 4);
  size_t arenaFloats = (size_t)(p - arena0) / 4;

  // ---- per-launch weight prep (deterministic) ----
  int n;
  n = VV * HH;   k_f32_to_bf16<<<(n + 255) / 256, 256, 0, stream>>>(trg_embed, trg_bf, n);
  n = G4 * 1024; k_pack_lstm<<<(n + 255) / 256, 256, 0, stream>>>(eWih[0], eWhh[0], Wc_e0, HH,   n);
  n = G4 * 1024; k_pack_lstm<<<(n + 255) / 256, 256, 0, stream>>>(eWih[1], eWhh[1], Wc_e1, HH,   n);
  n = G4 * 1536; k_pack_lstm<<<(n + 255) / 256, 256, 0, stream>>>(dWih[0], dWhh[0], Wc_d0, 1024, n);
  n = G4 * 1024; k_pack_lstm<<<(n + 255) / 256, 256, 0, stream>>>(dWih[1], dWhh[1], Wc_d1, HH,   n);
  n = G4; k_bsum<<<(n + 255) / 256, 256, 0, stream>>>(ebih[0], ebhh[0], bs_e0, n);
  n = G4; k_bsum<<<(n + 255) / 256, 256, 0, stream>>>(ebih[1], ebhh[1], bs_e1, n);
  n = G4; k_bsum<<<(n + 255) / 256, 256, 0, stream>>>(dbih[0], dbhh[0], bs_d0, n);
  n = G4; k_bsum<<<(n + 255) / 256, 256, 0, stream>>>(dbih[1], dbhh[1], bs_d1, n);
  n = HH * HH; k_f32_to_bf16<<<(n + 255) / 256, 256, 0, stream>>>(W1, W1bf, n);
  n = HH * HH; k_f32_to_bf16<<<(n + 255) / 256, 256, 0, stream>>>(W2, W2bf, n);
  n = LL * BB * HH;
  k_gather_embed<<<(n + 255) / 256, 256, 0, stream>>>(x, src_embed, A0, LL * BB, HH, HH);
  k_zero_f32<<<(unsigned)((arenaFloats + 255) / 256), 256, 0, stream>>>((float*)arena0, arenaFloats);

  // ---- encoder scan ----
  const int nCell = BB * HH;
  for (int t = 0; t < LL; ++t) {
    k_copy_rows_bf16<<<(nCell + 255) / 256, 256, 0, stream>>>(
        A0 + (size_t)t * BB * HH, HH, xh0, 1024, BB, HH);
    launch_gemm(xh0, 1024, Wc_e0, 1024, zbuf, G4, nullptr, BB, G4, 1024, stream);
    k_lstm_cell<<<(nCell + 255) / 256, 256, 0, stream>>>(
        zbuf, bs_e0, c_e0, xh0 + HH, 1024, xh1, 1024, nullptr, 0);
    launch_gemm(xh1, 1024, Wc_e1, 1024, zbuf, G4, nullptr, BB, G4, 1024, stream);
    k_lstm_cell<<<(nCell + 255) / 256, 256, 0, stream>>>(
        zbuf, bs_e1, c_e1, xh1 + HH, 1024,
        enc_obf + (size_t)t * BB * HH, HH, enc_o + (size_t)t * BB * HH, HH);
  }

  // ---- attention precompute: Wenc = enc_o @ W1^T + b1 ----
  launch_gemm(enc_obf, HH, W1bf, HH, Wenc, HH, b1, LL * BB, HH, HH, stream);

  // ---- decoder scan (teacher forced) ----
  for (int s = 0; s < TLm1; ++s) {
    k_gather_embed<<<(nCell + 255) / 256, 256, 0, stream>>>(
        y + s * BB, trg_embed, xhd0, BB, HH, 1536);
    launch_gemm(xhd1 + HH, 1024, W2bf, HH, qbuf, HH, b2, BB, HH, HH, stream);  // q = h_top@W2^T+b2
    k_attn<<<BB, 256, 0, stream>>>(Wenc, qbuf, enc_o, W3, b3, len, xhd0 + HH, 1536);
    launch_gemm(xhd0, 1536, Wc_d0, 1536, zbuf, G4, nullptr, BB, G4, 1536, stream);
    k_lstm_cell<<<(nCell + 255) / 256, 256, 0, stream>>>(
        zbuf, bs_d0, c_d0, xhd0 + 1024, 1536, xhd1, 1024, nullptr, 0);
    launch_gemm(xhd1, 1024, Wc_d1, 1024, zbuf, G4, nullptr, BB, G4, 1024, stream);
    k_lstm_cell<<<(nCell + 255) / 256, 256, 0, stream>>>(
        zbuf, bs_d1, c_d1, xhd1 + HH, 1024, nullptr, 0, nullptr, 0);
    // logits = h_top @ trg_embed^T + b_out  (dominant GEMM, B L2-resident)
    launch_gemm(xhd1 + HH, 1024, trg_bf, HH, out + (size_t)s * BB * VV, VV,
                b_out, BB, VV, HH, stream);
  }
}
